// GCNLayer_18073222381828
// MI455X (gfx1250) — matrix operand order repaired and checked by reading, not compiled, not run on hardware
//
#include <hip/hip_runtime.h>

typedef __attribute__((ext_vector_type(16))) __bf16       v16bf;
typedef __attribute__((ext_vector_type(8)))  float        v8f;
typedef __attribute__((ext_vector_type(4)))  unsigned int v4u;
typedef __attribute__((ext_vector_type(8)))  unsigned int v8u;

#define N_NODES 8192
#define D_IN    256
#define D_OUT   128

// ---------------------------------------------------------------------------
// TDM: issue a 2D tensor_load_to_lds (groups 0+1 only, VADDR2/3 = NULL form).
// D# packing per CDNA5 ISA 8.3/8.4. All inputs are block-uniform -> SGPRs.
//   data_size_code: 0=1B,1=2B,2=4B   pad codes per ISA table.
// ---------------------------------------------------------------------------
static __device__ __forceinline__ void tdm_load_2d(unsigned lds_off, const void* gptr,
                                                   unsigned tile_d0, unsigned tile_d1,
                                                   unsigned row_stride_elems,
                                                   unsigned data_size_code,
                                                   unsigned pad_interval_code,
                                                   unsigned pad_amount_code) {
    unsigned long long ga = (unsigned long long)(uintptr_t)gptr;
    unsigned g0a[4], g1a[8];
    g0a[0] = 0x1u;                                              // count=1, user mode
    g0a[1] = lds_off;                                           // lds_addr [63:32]
    g0a[2] = (unsigned)ga;                                      // global_addr lo
    g0a[3] = (unsigned)((ga >> 32) & 0x01FFFFFFu) | (2u << 30); // addr hi | type=2
    g1a[0] = (data_size_code << 16) | (1u << 20)                // pad_enable
           | (pad_interval_code << 22) | (pad_amount_code << 25);
    g1a[1] = (8192u & 0xFFFFu) << 16;                           // tensor_dim0 lo16 @ [63:48]
    g1a[2] = (8192u >> 16) | ((8192u & 0xFFFFu) << 16);         // dim0 hi16 | dim1 lo16
    g1a[3] = (8192u >> 16) | (tile_d0 << 16);                   // dim1 hi16 | tile_dim0
    g1a[4] = tile_d1;                                           // tile_dim1 | tile_dim2=0
    g1a[5] = row_stride_elems;                                  // tensor_dim0_stride lo32
    g1a[6] = 0;                                                 // stride hi | dim1_stride lo
    g1a[7] = 0;
    v4u g0; v8u g1;
    __builtin_memcpy(&g0, g0a, 16);
    __builtin_memcpy(&g1, g1a, 32);
    asm volatile("tensor_load_to_lds %0, %1" :: "s"(g0), "s"(g1) : "memory");
}

// ---------------------------------------------------------------------------
// Kernel 1: degree row-sums of adjacency -> dinv = rsqrt(sum + eps)
// ---------------------------------------------------------------------------
__global__ void k_degree(const float* __restrict__ A, float* __restrict__ dinv) {
    __shared__ float red[256];
    const int row = blockIdx.x;
    const float4* arow = (const float4*)(A + (size_t)row * N_NODES);
    float s = 0.f;
    #pragma unroll 4
    for (int j = threadIdx.x; j < N_NODES / 4; j += 256) {
        float4 v = arow[j];
        s += (v.x + v.y) + (v.z + v.w);
    }
    red[threadIdx.x] = s;
    __syncthreads();
    for (int off = 128; off > 0; off >>= 1) {
        if (threadIdx.x < off) red[threadIdx.x] += red[threadIdx.x + off];
        __syncthreads();
    }
    if (threadIdx.x == 0) dinv[row] = rsqrtf(red[0] + 1e-8f);
}

// ---------------------------------------------------------------------------
// Kernel 2: x = dinv * (values @ W + b), stored bf16 TRANSPOSED: xt[n][m]
// Native __bf16 casts -> v_cvt_pk_bf16_f32 class instructions.
// ---------------------------------------------------------------------------
__global__ void k_fc_scale(const float* __restrict__ V, const float* __restrict__ W,
                           const float* __restrict__ bias, const float* __restrict__ dinv,
                           unsigned short* __restrict__ xt) {
    const int lane = threadIdx.x & 31;
    const int wave = threadIdx.x >> 5;
    const int m0 = blockIdx.x * 16;
    const int n0 = wave * 16;
    const int half = lane >> 4;
    const int l16  = lane & 15;

    const int mA  = m0 + l16;
    const int kbA = half * 8;
    const int nB  = n0 + l16;
    const int kbB = half * 8;

    v8f acc = {};
    for (int k0 = 0; k0 < D_IN; k0 += 32) {
        const float* ap = V + (size_t)mA * D_IN + k0 + kbA;
        v16bf av, bv;
        #pragma unroll
        for (int j = 0; j < 8; ++j) av[j]     = (__bf16)ap[j];
        #pragma unroll
        for (int j = 0; j < 8; ++j) av[8 + j] = (__bf16)ap[16 + j];
        #pragma unroll
        for (int j = 0; j < 16; ++j)
            bv[j] = (__bf16)W[(size_t)(k0 + kbB + j + (j >= 8 ? 8 : 0)) * D_OUT + nB];
        acc = __builtin_amdgcn_wmma_f32_16x16x32_bf16(false, av, false, bv,
                                                      (short)0, acc, false, false);
    }

    // C layout: lane holds N = lane&15; VGPR r holds M = r + 8*(lane>>4).
    // Those 8 consecutive M's -> one 16B store into the transposed layout.
    const int mBase = m0 + half * 8;
    const float bn = bias[nB];
    __bf16 op[8];
    #pragma unroll
    for (int r = 0; r < 8; ++r)
        op[r] = (__bf16)(dinv[mBase + r] * (acc[r] + bn));
    uint4 q;
    __builtin_memcpy(&q, op, 16);
    *(uint4*)(xt + (size_t)nB * N_NODES + mBase) = q;
}

// ---------------------------------------------------------------------------
// Kernel 3: out = dinv_row * (A @ x).  M=8192, N=128, K=8192 (bf16 WMMA).
// Block = 16 rows x 128 cols, 8 waves (one 16x16 tile each), K-step 64.
// Double-buffered LDS fed by the Tensor Data Mover (wave 0 issues, TENSORcnt
// waited, workgroup barrier releases). TDM pad feature provides the LDS
// row padding (A: 64 DW rows + 4 DW pad = 68; X: 32 DW rows + 4 DW pad = 36).
// ---------------------------------------------------------------------------
__global__ void k_aggregate(const float* __restrict__ A, const unsigned short* __restrict__ xt,
                            const float* __restrict__ dinv, float* __restrict__ out) {
    __shared__ __attribute__((aligned(16))) float          Atile[2][16][68];   // fp32, pitch 272B
    __shared__ __attribute__((aligned(16))) unsigned short Xtile[2][128][72];  // bf16, pitch 144B

    const int t    = threadIdx.x;
    const int lane = t & 31;
    const int wave = t >> 5;
    const int m0   = blockIdx.x * 16;
    const int half = lane >> 4;
    const int l16  = lane & 15;
    const int n0   = wave * 16;
    const bool issuer = (wave == 0);      // uniform per wave -> scalar branch

    if (issuer) {
        tdm_load_2d((unsigned)(uintptr_t)&Atile[0][0][0],
                    A + (size_t)m0 * N_NODES, /*tile*/64, 16, N_NODES,
                    /*4B*/2, /*64 DW*/5, /*4 DW*/3);
        tdm_load_2d((unsigned)(uintptr_t)&Xtile[0][0][0],
                    xt, /*tile*/64, 128, N_NODES,
                    /*2B*/1, /*32 DW*/4, /*4 DW*/3);
    }

    v8f acc = {};
    const int NIT = N_NODES / 64;         // 128 iterations
    for (int kt = 0; kt < NIT; ++kt) {
        const int cur = kt & 1;
        if (issuer) __builtin_amdgcn_s_wait_tensorcnt(0);   // current tiles landed
        __syncthreads();                                     // release all waves; prev reads done
        if (issuer && (kt + 1 < NIT)) {                      // prefetch next tiles via TDM
            const int k0n = (kt + 1) * 64;
            tdm_load_2d((unsigned)(uintptr_t)&Atile[cur ^ 1][0][0],
                        A + (size_t)m0 * N_NODES + k0n, 64, 16, N_NODES, 2, 5, 3);
            tdm_load_2d((unsigned)(uintptr_t)&Xtile[cur ^ 1][0][0],
                        xt + k0n, 64, 128, N_NODES, 1, 4, 3);
        }
        #pragma unroll
        for (int ks = 0; ks < 64; ks += 32) {
            // A fragment: fp32 from LDS -> native bf16 cvt
            // (lane M = l16; K = kb..kb+15 contiguous; two ds_load_b128 pairs)
            const float* ar = &Atile[cur][l16][ks + half * 16];
            float af[16];
            __builtin_memcpy(af,     ar,      32);
            __builtin_memcpy(af + 8, ar + 8,  32);
            v16bf av;
            #pragma unroll
            for (int j = 0; j < 16; ++j) av[j] = (__bf16)af[j];
            // B fragment: bf16 direct (lane N = l16; K = kb'..kb'+15 contiguous)
            v16bf bv;
            __builtin_memcpy(&bv, &Xtile[cur][n0 + l16][ks + half * 16], 32);
            acc = __builtin_amdgcn_wmma_f32_16x16x32_bf16(false, av, false, bv,
                                                          (short)0, acc, false, false);
        }
    }

    const int mBase = m0 + half * 8;
    const int n = n0 + l16;
    #pragma unroll
    for (int r = 0; r < 8; ++r)
        out[(size_t)(mBase + r) * D_OUT + n] = dinv[mBase + r] * acc[r];
}

// ---------------------------------------------------------------------------
extern "C" void kernel_launch(void* const* d_in, const int* in_sizes, int n_in,
                              void* d_out, int out_size, void* d_ws, size_t ws_size,
                              hipStream_t stream) {
    const float* values    = (const float*)d_in[0];   // [8192, 256]
    const float* adjacency = (const float*)d_in[1];   // [8192, 8192]
    const float* W         = (const float*)d_in[2];   // [256, 128]
    const float* bias      = (const float*)d_in[3];   // [128]
    float* out = (float*)d_out;                        // [8192, 128]

    float*          dinv = (float*)d_ws;                              // 32 KB
    unsigned short* xt   = (unsigned short*)((char*)d_ws + 32768);    // bf16 [128][8192], 2 MB

    k_degree   <<<dim3(N_NODES),      dim3(256), 0, stream>>>(adjacency, dinv);
    k_fc_scale <<<dim3(N_NODES / 16), dim3(256), 0, stream>>>(values, W, bias, dinv, xt);
    k_aggregate<<<dim3(N_NODES / 16), dim3(256), 0, stream>>>(adjacency, xt, dinv, out);
}
